// MultiHeadAttention_9869834846801
// MI455X (gfx1250) — matrix operand order repaired and checked by reading, not compiled, not run on hardware
//
#include <hip/hip_runtime.h>
#include <hip/hip_fp16.h>

typedef _Float16 hv2  __attribute__((ext_vector_type(2)));
typedef _Float16 hv8  __attribute__((ext_vector_type(8)));
typedef _Float16 hv16 __attribute__((ext_vector_type(16)));
typedef float    fv8  __attribute__((ext_vector_type(8)));

#if defined(__AMDGCN__) && __has_builtin(__builtin_amdgcn_s_wait_asynccnt)
#define HAVE_ASYNC_LDS 1
// Async global->LDS copy (16B per lane), tracked by ASYNCcnt.
// VDST = VGPR holding LDS byte offset (generic shared ptr truncated to 32b,
// per ISA flat-addressing rule: LDS address is the low 32 bits); VADDR = 64-bit global.
static __device__ __forceinline__ void async_load_lds_b128(const _Float16* g,
                                                           _Float16* l) {
  unsigned loff = (unsigned)(unsigned long long)l;
  unsigned long long ga = (unsigned long long)g;
  asm volatile("global_load_async_to_lds_b128 %0, %1, off"
               :
               : "v"(loff), "v"(ga)
               : "memory");
}
#else
#define HAVE_ASYNC_LDS 0
#endif

// XOR-16 lane swap via ds_swizzle (bitmask mode: and=0x1f, or=0, xor=0x10)
static __device__ __forceinline__ float swz_xor16(float x) {
  return __int_as_float(__builtin_amdgcn_ds_swizzle(__float_as_int(x), 0x401f));
}

static __device__ __forceinline__ fv8 wmma_f16(hv16 a, hv16 b, fv8 c) {
  return __builtin_amdgcn_wmma_f32_16x16x32_f16(false, a, false, b, (short)0, c,
                                                false, false);
}

// ---------------------------------------------------------------------------
// GEMM: Y = X(MxK) @ W(KxN) + bias, via v_wmma_f32_16x16x32_f16.
// Block: 256 threads = 8 waves. Block tile 128(M) x 64(N); each wave owns
// 16 rows x 64 cols (4 accumulator tiles) -> one A-fragment feeds 4 WMMAs.
// The 32x64 W k-slab is staged transposed (f16, [n][k]) in LDS and shared by
// all 8 waves.
// MODE 0: f16 out, row-major (Q/K projections)
// MODE 1: f16 out, per-head transposed to (b, h, dh, T)  (V projection)
// MODE 2: f32 out, row-major (final output projection)
// ---------------------------------------------------------------------------
template <typename TIN, int MODE>
__global__ __launch_bounds__(256) void gemm_wmma_kernel(
    const TIN* __restrict__ X, const float* __restrict__ W,
    const float* __restrict__ bias, void* __restrict__ Yv,
    int M, int N, int K, int T) {
  __shared__ _Float16 ldsWT[64 * 32];  // [n][k]

  const int tid   = threadIdx.x;
  const int lane  = tid & 31;
  const int col   = lane & 15;
  const int khalf = lane >> 4;
  const int wid   = tid >> 5;
  const int n0    = blockIdx.x * 64;
  const int m0    = blockIdx.y * 128 + wid * 16;

  fv8 acc[4];
#pragma unroll
  for (int nt = 0; nt < 4; ++nt)
#pragma unroll
    for (int i = 0; i < 8; ++i) acc[nt][i] = 0.0f;

  // cooperative W staging: thread -> col n0+sn, k rows sk..sk+7
  const int sn = tid & 63;
  const int sk = (tid >> 6) * 8;

  const size_t arow = (size_t)(m0 + col) * (size_t)K;

  for (int k0 = 0; k0 < K; k0 += 32) {
    // stage W^T slab: 32(K) x 64(N) fp32 -> f16 LDS [n][k]
    hv8 wv;
#pragma unroll
    for (int i = 0; i < 8; ++i)
      wv[i] = (_Float16)W[(size_t)(k0 + sk + i) * N + n0 + sn];
    *reinterpret_cast<hv8*>(&ldsWT[sn * 32 + sk]) = wv;

    // prefetch next slab / next A chunk while waves sync
    if (k0 + 32 < K) {
      __builtin_prefetch(&W[(size_t)(k0 + 32 + sk) * N + n0 + sn], 0, 1);
      __builtin_prefetch((const TIN*)X + arow + k0 + 32, 0, 1);
    }
    __syncthreads();

    // A fragment: lane row = m0+col, elements k = khalf*8+0..7, 16+khalf*8+0..7
    hv16 af;
    if constexpr (sizeof(TIN) == 4) {
      const float* xr = (const float*)X + arow + k0;
      float4 f0 = *reinterpret_cast<const float4*>(xr + khalf * 8);
      float4 f1 = *reinterpret_cast<const float4*>(xr + khalf * 8 + 4);
      float4 f2 = *reinterpret_cast<const float4*>(xr + 16 + khalf * 8);
      float4 f3 = *reinterpret_cast<const float4*>(xr + 16 + khalf * 8 + 4);
      af[0] = (_Float16)f0.x;  af[1] = (_Float16)f0.y;
      af[2] = (_Float16)f0.z;  af[3] = (_Float16)f0.w;
      af[4] = (_Float16)f1.x;  af[5] = (_Float16)f1.y;
      af[6] = (_Float16)f1.z;  af[7] = (_Float16)f1.w;
      af[8] = (_Float16)f2.x;  af[9] = (_Float16)f2.y;
      af[10] = (_Float16)f2.z; af[11] = (_Float16)f2.w;
      af[12] = (_Float16)f3.x; af[13] = (_Float16)f3.y;
      af[14] = (_Float16)f3.z; af[15] = (_Float16)f3.w;
    } else {
      const _Float16* xr = (const _Float16*)X + arow + k0;
      hv8 lo = *reinterpret_cast<const hv8*>(xr + khalf * 8);
      hv8 hi = *reinterpret_cast<const hv8*>(xr + 16 + khalf * 8);
#pragma unroll
      for (int i = 0; i < 8; ++i) { af[i] = lo[i]; af[8 + i] = hi[i]; }
    }

    // 4 B fragments from LDS (two 16B pieces each, k = khalf*8+0..7, 16+khalf*8+0..7), 4 WMMAs
#pragma unroll
    for (int nt = 0; nt < 4; ++nt) {
      const _Float16* wp = &ldsWT[(nt * 16 + col) * 32 + khalf * 8];
      hv8 blo = *reinterpret_cast<const hv8*>(wp);
      hv8 bhi = *reinterpret_cast<const hv8*>(wp + 16);
      hv16 bf;
#pragma unroll
      for (int i = 0; i < 8; ++i) { bf[i] = blo[i]; bf[8 + i] = bhi[i]; }
      acc[nt] = wmma_f16(af, bf, acc[nt]);
    }
    __syncthreads();
  }

  // Epilogue: C layout -> lane holds column nt*16+col, rows m0 + khalf*8 + r
#pragma unroll
  for (int nt = 0; nt < 4; ++nt) {
    const int colg = n0 + nt * 16 + col;
    const float bv = bias[colg];
    if constexpr (MODE == 0) {
      _Float16* Y = (_Float16*)Yv;
#pragma unroll
      for (int r = 0; r < 8; ++r) {
        int row = m0 + khalf * 8 + r;
        Y[(size_t)row * N + colg] = (_Float16)(acc[nt][r] + bv);
      }
    } else if constexpr (MODE == 1) {
      _Float16* Y = (_Float16*)Yv;
      const int hh = colg >> 6;
      const int d  = colg & 63;
      const int heads = N >> 6;
#pragma unroll
      for (int r = 0; r < 8; ++r) {
        int row = m0 + khalf * 8 + r;
        int bi = row / T;
        int t  = row - bi * T;
        Y[((size_t)(bi * heads + hh) * 64 + d) * (size_t)T + t] =
            (_Float16)(acc[nt][r] + bv);
      }
    } else {
      float* Y = (float*)Yv;
#pragma unroll
      for (int r = 0; r < 8; ++r) {
        int row = m0 + khalf * 8 + r;
        Y[(size_t)row * N + colg] = acc[nt][r] + bv;
      }
    }
  }
}

// ---------------------------------------------------------------------------
// Flash attention. Block = 4 waves, all on the same (b,h); each wave owns two
// 16-query tiles (32 queries). Per 32-key block, the K tile (32x64 f16) and
// V^T tile (64x32 f16) are staged into LDS ONCE per block (gfx1250
// global_load_async_to_lds_b128 + s_wait_asynccnt when available; synchronous
// staging fallback otherwise) and shared by all 4 waves -> 4x less global
// traffic. Scores computed as S^T = K @ Q^T (lane = query column -> per-lane
// softmax stats + one XOR-16 ds_swizzle). Output accumulated as O^T = V^T@P^T
// with V stored (b,h,dh,T) so all staging/fragment accesses are contiguous.
// ---------------------------------------------------------------------------
__global__ __launch_bounds__(128) void attn_kernel(
    const _Float16* __restrict__ Qh, const _Float16* __restrict__ Kh,
    const _Float16* __restrict__ Vt, const int* __restrict__ mask,
    _Float16* __restrict__ O, int T, int D, int Hn, float scale) {
  __shared__ _Float16 ldsK[32 * 64];  // [key][dh]
  __shared__ _Float16 ldsV[64 * 32];  // [dh][key]

  const int lane  = threadIdx.x & 31;
  const int wid   = threadIdx.x >> 5;
  const int col   = lane & 15;
  const int khalf = lane >> 4;
  const int bh = blockIdx.y;
  const int b  = bh / Hn;
  const int h  = bh - b * Hn;
  const int q0 = (blockIdx.x * 4 + wid) * 32;

  const size_t vbase = (size_t)((b * Hn + h) * 64) * (size_t)T;

  // staging coordinates: waves 0-1 stage K (256 16B chunks), waves 2-3 stage V
  // K chunk c: row = c>>3 (key), half-off = (c&7)*8 ; V chunk c: row = c>>2 (dh)
  const int kc_base = wid * 128 + lane;          // used when wid<2
  const int vc_base = (wid - 2) * 128 + lane;    // used when wid>=2

  // Q^T B-fragments (loop invariant), two query tiles
  hv16 qf[2][2];
#pragma unroll
  for (int qt = 0; qt < 2; ++qt) {
    const _Float16* qp =
        Qh + ((size_t)(b * T + q0 + qt * 16 + col)) * D + h * 64;
#pragma unroll
    for (int kc = 0; kc < 2; ++kc) {
      hv8 lo = *reinterpret_cast<const hv8*>(qp + kc * 32 + khalf * 8);
      hv8 hi = *reinterpret_cast<const hv8*>(qp + kc * 32 + 16 + khalf * 8);
#pragma unroll
      for (int i = 0; i < 8; ++i) {
        qf[qt][kc][i] = lo[i];
        qf[qt][kc][8 + i] = hi[i];
      }
    }
  }

  fv8 oacc[2][4];
#pragma unroll
  for (int qt = 0; qt < 2; ++qt)
#pragma unroll
    for (int t = 0; t < 4; ++t)
#pragma unroll
      for (int i = 0; i < 8; ++i) oacc[qt][t][i] = 0.0f;

  float m_run[2] = {-1e30f, -1e30f};
  float l_run[2] = {0.0f, 0.0f};

  for (int key0 = 0; key0 < T; key0 += 32) {
    // ---- stage K (32x64) and V^T (64x32) tiles into LDS ----
    if (wid < 2) {
#pragma unroll
      for (int t = 0; t < 4; ++t) {
        int c = kc_base + t * 32;         // 0..255
        int row = c >> 3;                 // key row 0..31
        int off = (c & 7) * 8;            // half offset in row (0..56)
        const _Float16* g =
            Kh + ((size_t)(b * T + key0 + row)) * D + h * 64 + off;
        _Float16* l = &ldsK[row * 64 + off];
#if HAVE_ASYNC_LDS
        async_load_lds_b128(g, l);
#else
        *reinterpret_cast<hv8*>(l) = *reinterpret_cast<const hv8*>(g);
#endif
      }
      if (key0 + 32 < T)
        __builtin_prefetch(
            Kh + ((size_t)(b * T + key0 + 32 + (kc_base >> 3))) * D + h * 64, 0,
            1);
    } else {
#pragma unroll
      for (int t = 0; t < 4; ++t) {
        int c = vc_base + t * 32;         // 0..255
        int row = c >> 2;                 // dh row 0..63
        int off = (c & 3) * 8;            // half offset in row (0..24)
        const _Float16* g = Vt + vbase + (size_t)row * T + key0 + off;
        _Float16* l = &ldsV[row * 32 + off];
#if HAVE_ASYNC_LDS
        async_load_lds_b128(g, l);
#else
        *reinterpret_cast<hv8*>(l) = *reinterpret_cast<const hv8*>(g);
#endif
      }
      if (key0 + 32 < T)
        __builtin_prefetch(Vt + vbase + (size_t)(vc_base >> 2) * T + key0 + 32,
                           0, 1);
    }
#if HAVE_ASYNC_LDS
    __builtin_amdgcn_s_wait_asynccnt(0);
#endif
    __syncthreads();

    // ---- scores: S^T tiles in C layout; K-fragments from LDS ----
    fv8 s[2][2];
#pragma unroll
    for (int qt = 0; qt < 2; ++qt)
#pragma unroll
      for (int st = 0; st < 2; ++st)
#pragma unroll
        for (int i = 0; i < 8; ++i) s[qt][st][i] = 0.0f;

#pragma unroll
    for (int st = 0; st < 2; ++st) {
      const _Float16* kp = &ldsK[(st * 16 + col) * 64];
#pragma unroll
      for (int kc = 0; kc < 2; ++kc) {
        hv16 kf;
        hv8 lo = *reinterpret_cast<const hv8*>(kp + kc * 32 + khalf * 8);
        hv8 hi = *reinterpret_cast<const hv8*>(kp + kc * 32 + 16 + khalf * 8);
#pragma unroll
        for (int i = 0; i < 8; ++i) { kf[i] = lo[i]; kf[8 + i] = hi[i]; }
        s[0][st] = wmma_f16(kf, qf[0][kc], s[0][st]);
        s[1][st] = wmma_f16(kf, qf[1][kc], s[1][st]);
      }
    }

    // ---- per-key mask bits for this lane's 16 keys (shared by q-tiles) ----
    const int* mrow = mask + b * T + key0;
    int4 ma = *reinterpret_cast<const int4*>(mrow + khalf * 8);
    int4 mb = *reinterpret_cast<const int4*>(mrow + khalf * 8 + 4);
    int4 mc = *reinterpret_cast<const int4*>(mrow + 16 + khalf * 8);
    int4 md = *reinterpret_cast<const int4*>(mrow + 16 + khalf * 8 + 4);
    int ks0[8] = {ma.x, ma.y, ma.z, ma.w, mb.x, mb.y, mb.z, mb.w};
    int ks1[8] = {mc.x, mc.y, mc.z, mc.w, md.x, md.y, md.z, md.w};

    // ---- online softmax per query tile; build P^T B-fragments ----
    hv16 pf[2];
#pragma unroll
    for (int qt = 0; qt < 2; ++qt) {
      float bm = -1e30f;
      float sc0[8], sc1[8];
#pragma unroll
      for (int r = 0; r < 8; ++r) {
        sc0[r] = s[qt][0][r] * scale;
        sc1[r] = s[qt][1][r] * scale;
        bm = ks0[r] ? fmaxf(bm, sc0[r]) : bm;
        bm = ks1[r] ? fmaxf(bm, sc1[r]) : bm;
      }
      bm = fmaxf(bm, swz_xor16(bm));
      float m_new = fmaxf(m_run[qt], bm);
      float alpha = __expf(m_run[qt] - m_new);

      float p0[8], p1[8];
      float bs = 0.0f;
#pragma unroll
      for (int r = 0; r < 8; ++r) {
        p0[r] = ks0[r] ? __expf(sc0[r] - m_new) : 0.0f;
        p1[r] = ks1[r] ? __expf(sc1[r] - m_new) : 0.0f;
        bs += p0[r] + p1[r];
      }
      bs += swz_xor16(bs);
      l_run[qt] = l_run[qt] * alpha + bs;
      m_run[qt] = m_new;
#pragma unroll
      for (int t = 0; t < 4; ++t)
#pragma unroll
        for (int i = 0; i < 8; ++i) oacc[qt][t][i] *= alpha;

#pragma unroll
      for (int r = 0; r < 8; ++r) {
        float xsend = (khalf == 0) ? p1[r] : p0[r];
        float y = swz_xor16(xsend);
        pf[qt][r]     = (_Float16)((khalf == 0) ? p0[r] : y);
        pf[qt][8 + r] = (_Float16)((khalf == 0) ? y : p1[r]);
      }
    }

    // ---- O^T += V^T @ P^T: V fragments from LDS, shared by both q-tiles ----
#pragma unroll
    for (int t = 0; t < 4; ++t) {
      const _Float16* vp = &ldsV[(t * 16 + col) * 32];
      hv16 vf;
      hv8 lo = *reinterpret_cast<const hv8*>(vp + khalf * 16);
      hv8 hi = *reinterpret_cast<const hv8*>(vp + khalf * 16 + 8);
#pragma unroll
      for (int i = 0; i < 8; ++i) { vf[i] = lo[i]; vf[8 + i] = hi[i]; }
      oacc[0][t] = wmma_f16(vf, pf[0], oacc[0][t]);
      oacc[1][t] = wmma_f16(vf, pf[1], oacc[1][t]);
    }
    __syncthreads();  // LDS tiles fully consumed before next staging
  }

  // ---- normalize + store (8 contiguous dh per lane per tile) ----
#pragma unroll
  for (int qt = 0; qt < 2; ++qt) {
    float inv = (l_run[qt] > 0.0f) ? 1.0f / l_run[qt] : 0.0f;
    _Float16* op =
        O + ((size_t)(b * T + q0 + qt * 16 + col)) * D + h * 64 + khalf * 8;
#pragma unroll
    for (int t = 0; t < 4; ++t) {
      hv8 ov;
#pragma unroll
      for (int r = 0; r < 8; ++r) ov[r] = (_Float16)(oacc[qt][t][r] * inv);
      *reinterpret_cast<hv8*>(op + t * 16) = ov;
    }
  }
}

// ---------------------------------------------------------------------------
extern "C" void kernel_launch(void* const* d_in, const int* in_sizes, int n_in,
                              void* d_out, int out_size, void* d_ws,
                              size_t ws_size, hipStream_t stream) {
  const int Bn = 2, T = 2048, D = 1024, H = 16;
  const int M = Bn * T;

  const float* q   = (const float*)d_in[0];
  const float* k   = (const float*)d_in[1];
  const float* v   = (const float*)d_in[2];
  const int* mask  = (const int*)d_in[3];
  const float* Wq  = (const float*)d_in[4];
  const float* bq  = (const float*)d_in[5];
  const float* Wk  = (const float*)d_in[6];
  const float* bk  = (const float*)d_in[7];
  const float* Wv  = (const float*)d_in[8];
  const float* bv  = (const float*)d_in[9];
  const float* Wo  = (const float*)d_in[10];
  const float* bo  = (const float*)d_in[11];
  float* out = (float*)d_out;

  // workspace: qh | kh | vt | oh, each M*D f16 (8 MB) => 32 MB total
  _Float16* qh = (_Float16*)d_ws;
  _Float16* kh = qh + (size_t)M * D;
  _Float16* vt = kh + (size_t)M * D;
  _Float16* oh = vt + (size_t)M * D;

  dim3 ggrid(D / 64, M / 128);
  gemm_wmma_kernel<float, 0><<<ggrid, 256, 0, stream>>>(q, Wq, bq, qh, M, D, D, T);
  gemm_wmma_kernel<float, 0><<<ggrid, 256, 0, stream>>>(k, Wk, bk, kh, M, D, D, T);
  gemm_wmma_kernel<float, 1><<<ggrid, 256, 0, stream>>>(v, Wv, bv, vt, M, D, D, T);

  dim3 agrid(T / 128, Bn * H);
  // NOTE: reference scales by 1/sqrt(D) with FULL D=1024 -> 1/32
  attn_kernel<<<agrid, 128, 0, stream>>>(qh, kh, vt, mask, oh, T, D, H,
                                         0.03125f);

  gemm_wmma_kernel<_Float16, 2><<<ggrid, 256, 0, stream>>>(oh, Wo, bo, out, M,
                                                           D, D, T);
}
